// CombinedGNNModel_55817394978868
// MI455X (gfx1250) — matrix operand order repaired and checked by reading, not compiled, not run on hardware
//
#include <hip/hip_runtime.h>
#include <hip/hip_bf16.h>
#include <math.h>

#define NN 50000
#define EE 800000
#define DD 128
#define HH 128
#define NHEADS 8
#define HDIM 16
#define CC 64
#define NEG_SLOPE 0.2f

typedef __attribute__((ext_vector_type(16))) __bf16 v16bf;
typedef __attribute__((ext_vector_type(8)))  float  v8f;

__device__ __forceinline__ __bf16 f2bf(float f) {
    unsigned u = __float_as_uint(f);
    unsigned r = (u + 0x7FFFu + ((u >> 16) & 1u)) >> 16;   // RNE
    unsigned short s = (unsigned short)r;
    return __builtin_bit_cast(__bf16, s);
}

// order-preserving float<->uint mapping for atomicMax on floats
__device__ __forceinline__ unsigned fmapu(float f) {
    unsigned b = __float_as_uint(f);
    return (b & 0x80000000u) ? ~b : (b | 0x80000000u);
}
__device__ __forceinline__ float funmap(unsigned u) {
    unsigned b = (u & 0x80000000u) ? (u & 0x7FFFFFFFu) : ~u;
    return __uint_as_float(b);
}

// ---------------------------------------------------------------------------
// Weight pre-pack: Wt[n][k] = bf16( W[k][n] )   (K = 128 fixed)
// Row-major fp32 W[K,Nout] -> transposed bf16 Wt[Nout,128] so that the WMMA
// B-fragment (16 consecutive K values per lane) is one contiguous 32B load.
// ---------------------------------------------------------------------------
__global__ void pack_w_kernel(const float* __restrict__ W,
                              unsigned short* __restrict__ Wt, int Nout)
{
    int i = blockIdx.x * blockDim.x + threadIdx.x;     // Nout*128
    if (i >= Nout * 128) return;
    int n = i >> 7, k = i & 127;
    unsigned u = __float_as_uint(W[(size_t)k * Nout + n]);
    unsigned r = (u + 0x7FFFu + ((u >> 16) & 1u)) >> 16;
    Wt[i] = (unsigned short)r;
}

// ---------------------------------------------------------------------------
// WMMA GEMM: C[M,Nout] = act( rowscale(A[M,128]) @ W[128,Nout] + bias )
// One wave -> one 16x16 output tile; K = 128 -> 4x v_wmma_f32_16x16x32_bf16.
// A fragment: two contiguous 8-float runs per K-step -> 4x b128 loads.
// B fragment: 16 contiguous bf16 from packed Wt -> 2x b128 loads.
// ---------------------------------------------------------------------------
__global__ void wmma_gemm_kernel(const float* __restrict__ A,
                                 const unsigned short* __restrict__ Wt,
                                 const float* __restrict__ bias,
                                 const float* __restrict__ rowScaleDeg,
                                 float* __restrict__ Cout,
                                 int M, int Nout, int act)
{
    const int lane = threadIdx.x & 31;
    const int wave = threadIdx.x >> 5;
    const int tilesN = Nout >> 4;
    const int totalTiles = (M >> 4) * tilesN;
    const int t = blockIdx.x * (blockDim.x >> 5) + wave;
    if (t >= totalTiles) return;                 // wave-uniform exit
    const int mt = t / tilesN, nt = t % tilesN;
    const int m0 = mt << 4, n0 = nt << 4;
    const int mn = lane & 15;                    // row for A, col for B/C
    const int g  = lane >> 4;

    const float* Arow = A + (size_t)(m0 + mn) * 128;
    float scale = 1.0f;
    if (rowScaleDeg) scale = 1.0f / fmaxf(rowScaleDeg[m0 + mn], 1.0f);
    const unsigned short* Brow = Wt + (size_t)(n0 + mn) * 128;

    v8f c = {};
    #pragma unroll
    for (int kk = 0; kk < 128; kk += 32) {
        // A 16x32 bf16 fragment: lane (m=mn, g): K = g*16 + [0..15], contiguous
        const float4* pa0 = (const float4*)(Arow + kk + g * 16);
        const float4* pa1 = (const float4*)(Arow + kk + g * 16 + 8);
        float4 a0 = pa0[0], a1 = pa0[1];
        float4 a2 = pa1[0], a3 = pa1[1];
        v16bf a;
        a[0]  = f2bf(a0.x * scale); a[1]  = f2bf(a0.y * scale);
        a[2]  = f2bf(a0.z * scale); a[3]  = f2bf(a0.w * scale);
        a[4]  = f2bf(a1.x * scale); a[5]  = f2bf(a1.y * scale);
        a[6]  = f2bf(a1.z * scale); a[7]  = f2bf(a1.w * scale);
        a[8]  = f2bf(a2.x * scale); a[9]  = f2bf(a2.y * scale);
        a[10] = f2bf(a2.z * scale); a[11] = f2bf(a2.w * scale);
        a[12] = f2bf(a3.x * scale); a[13] = f2bf(a3.y * scale);
        a[14] = f2bf(a3.z * scale); a[15] = f2bf(a3.w * scale);
        // B 32x16 bf16 fragment: lane (n=mn, g): K = g*16 + [0..15], contiguous
        v16bf b = *(const v16bf*)(Brow + kk + g * 16);
        c = __builtin_amdgcn_wmma_f32_16x16x32_bf16(false, a, false, b,
                                                    (short)0, c, false, false);
    }

    const float bv = bias ? bias[n0 + mn] : 0.0f;
    #pragma unroll
    for (int v = 0; v < 8; ++v) {
        int row = m0 + v + g * 8;                // C/D layout: M = v + 8*(lane>=16)
        float o = c[v] + bv;
        if (act == 1) o = fmaxf(o, 0.0f);
        Cout[(size_t)row * Nout + n0 + mn] = o;
    }
}

// ---------------------------------------------------------------------------
// Edge / segment kernels
// ---------------------------------------------------------------------------
__global__ void deg_kernel(const int* __restrict__ dst, float* __restrict__ deg, int E)
{
    int e = blockIdx.x * blockDim.x + threadIdx.x;
    if (e < E) atomicAdd(&deg[dst[e]], 1.0f);
}

// agg[dst[e], f] += X[src[e], f]   (F = 128)
__global__ void scatter_feat_kernel(const float* __restrict__ X,
                                    const int* __restrict__ src,
                                    const int* __restrict__ dst,
                                    float* __restrict__ agg, int E)
{
    int i = blockIdx.x * blockDim.x + threadIdx.x;
    int e = i >> 7, f = i & 127;
    if (e >= E) return;
    atomicAdd(&agg[(size_t)dst[e] * 128 + f], X[(size_t)src[e] * 128 + f]);
}

__global__ void gat_es_kernel(const float* __restrict__ h,
                              const float* __restrict__ a_src,
                              const float* __restrict__ a_dst,
                              float* __restrict__ es, float* __restrict__ ed, int Nn)
{
    int i = blockIdx.x * blockDim.x + threadIdx.x;
    if (i >= Nn * NHEADS) return;
    int n = i >> 3, hh = i & 7;
    const float* hp = h + (size_t)n * HH + hh * HDIM;
    const float* as = a_src + hh * HDIM;
    const float* ad = a_dst + hh * HDIM;
    float s = 0.f, d2 = 0.f;
    #pragma unroll
    for (int j = 0; j < HDIM; ++j) { float v = hp[j]; s += v * as[j]; d2 += v * ad[j]; }
    es[i] = s; ed[i] = d2;
}

__device__ __forceinline__ float edge_score(const float* es, const float* ed,
                                            int s, int d, int hh)
{
    float sc = es[(size_t)s * NHEADS + hh] + ed[(size_t)d * NHEADS + hh];
    return sc > 0.f ? sc : sc * NEG_SLOPE;
}

__global__ void gat_max_kernel(const float* __restrict__ es, const float* __restrict__ ed,
                               const int* __restrict__ src, const int* __restrict__ dst,
                               unsigned* __restrict__ mm, int E)
{
    int i = blockIdx.x * blockDim.x + threadIdx.x;
    if (i >= E * NHEADS) return;
    int e = i >> 3, hh = i & 7;
    int d = dst[e];
    float sc = edge_score(es, ed, src[e], d, hh);
    atomicMax(&mm[(size_t)d * NHEADS + hh], fmapu(sc));
}

__global__ void gat_exp_kernel(const float* __restrict__ es, const float* __restrict__ ed,
                               const int* __restrict__ src, const int* __restrict__ dst,
                               const unsigned* __restrict__ mm,
                               float* __restrict__ exb, float* __restrict__ ssum, int E)
{
    int i = blockIdx.x * blockDim.x + threadIdx.x;
    if (i >= E * NHEADS) return;
    int e = i >> 3, hh = i & 7;
    int d = dst[e];
    float sc = edge_score(es, ed, src[e], d, hh);
    float m = funmap(mm[(size_t)d * NHEADS + hh]);
    float ex = expf(sc - m);
    exb[i] = ex;
    atomicAdd(&ssum[(size_t)d * NHEADS + hh], ex);
}

// out[dst, f] += h[src, f] * ex[e, f/16] / (ssum[dst, f/16] + 1e-9)
__global__ void gat_scatter_kernel(const float* __restrict__ h,
                                   const int* __restrict__ src,
                                   const int* __restrict__ dst,
                                   const float* __restrict__ exb,
                                   const float* __restrict__ ssum,
                                   float* __restrict__ out, int E)
{
    int i = blockIdx.x * blockDim.x + threadIdx.x;
    int e = i >> 7, f = i & 127;
    if (e >= E) return;
    int hh = f >> 4;
    int d = dst[e];
    float w = exb[(size_t)e * NHEADS + hh] / (ssum[(size_t)d * NHEADS + hh] + 1e-9f);
    atomicAdd(&out[(size_t)d * HH + f], h[(size_t)src[e] * HH + f] * w);
}

// ---------------------------------------------------------------------------
// Row softmax over 64 columns, in place. One wave per row (wave32).
// ---------------------------------------------------------------------------
__global__ void softmax64_kernel(float* __restrict__ X, int M)
{
    int lane = threadIdx.x & 31;
    int row = blockIdx.x * (blockDim.x >> 5) + (threadIdx.x >> 5);
    if (row >= M) return;
    float* p = X + (size_t)row * CC;
    float a = p[lane * 2], b = p[lane * 2 + 1];
    float m = fmaxf(a, b);
    #pragma unroll
    for (int off = 16; off; off >>= 1) m = fmaxf(m, __shfl_xor(m, off));
    a = expf(a - m); b = expf(b - m);
    float s = a + b;
    #pragma unroll
    for (int off = 16; off; off >>= 1) s += __shfl_xor(s, off);
    float inv = 1.0f / s;
    p[lane * 2] = a * inv; p[lane * 2 + 1] = b * inv;
}

// y[n] = sigmoid( sum_f relu(out[n,f]+bg[f]) * Wl[f] + bl ). One wave per row.
__global__ void link_kernel(const float* __restrict__ out, const float* __restrict__ bg,
                            const float* __restrict__ Wl, const float* __restrict__ bl,
                            float* __restrict__ y, int M)
{
    int lane = threadIdx.x & 31;
    int row = blockIdx.x * (blockDim.x >> 5) + (threadIdx.x >> 5);
    if (row >= M) return;
    float4 v  = ((const float4*)(out + (size_t)row * HH))[lane];
    float4 g4 = ((const float4*)bg)[lane];
    float4 w4 = ((const float4*)Wl)[lane];
    float acc = fmaxf(v.x + g4.x, 0.f) * w4.x
              + fmaxf(v.y + g4.y, 0.f) * w4.y
              + fmaxf(v.z + g4.z, 0.f) * w4.z
              + fmaxf(v.w + g4.w, 0.f) * w4.w;
    #pragma unroll
    for (int off = 16; off; off >>= 1) acc += __shfl_xor(acc, off);
    if (lane == 0) y[row] = 1.0f / (1.0f + expf(-(acc + bl[0])));
}

// ---------------------------------------------------------------------------
extern "C" void kernel_launch(void* const* d_in, const int* in_sizes, int n_in,
                              void* d_out, int out_size, void* d_ws, size_t ws_size,
                              hipStream_t stream)
{
    const float* x     = (const float*)d_in[0];
    const int*   ei    = (const int*)d_in[1];
    const float* W1    = (const float*)d_in[2];
    const float* b1    = (const float*)d_in[3];
    const float* W2    = (const float*)d_in[4];
    const float* b2    = (const float*)d_in[5];
    const float* Wg    = (const float*)d_in[6];
    const float* bg    = (const float*)d_in[7];
    const float* a_src = (const float*)d_in[8];
    const float* a_dst = (const float*)d_in[9];
    const float* Wcls  = (const float*)d_in[10];
    const float* bcls  = (const float*)d_in[11];
    const float* Wl    = (const float*)d_in[12];
    const float* bl    = (const float*)d_in[13];

    const int E = in_sizes[1] / 2;
    const int* src = ei;
    const int* dst = ei + E;

    float* ws   = (float*)d_ws;
    float* deg  = ws;                                  // N
    float* tmp  = deg + NN;                            // N*128 (agg / gat-out)
    float* xb   = tmp + (size_t)NN * HH;               // N*128 (x1, then h)
    float* hb   = xb  + (size_t)NN * HH;               // N*128 (x2)
    float* es   = hb  + (size_t)NN * HH;               // N*8
    float* ed   = es  + (size_t)NN * NHEADS;           // N*8
    unsigned* mm = (unsigned*)(ed + (size_t)NN * NHEADS); // N*8
    float* ssum = (float*)(mm + (size_t)NN * NHEADS);  // N*8
    float* exb  = ssum + (size_t)NN * NHEADS;          // E*8
    unsigned short* Wt1 = (unsigned short*)(exb + (size_t)E * NHEADS); // 128*128
    unsigned short* Wt2 = Wt1 + 128 * 128;             // 128*128
    unsigned short* Wtg = Wt2 + 128 * 128;             // 128*128
    unsigned short* Wtc = Wtg + 128 * 128;             // 64*128

    float* logits = (float*)d_out;                     // [N,64]
    float* linkp  = (float*)d_out + (size_t)NN * CC;   // [N,1]

    auto gemm = [&](const float* A, const unsigned short* Wt, const float* bias,
                    const float* degp, float* C, int Nout, int act) {
        int tiles = (NN / 16) * (Nout / 16);
        int blocks = (tiles + 7) / 8;                  // 8 waves / block
        wmma_gemm_kernel<<<blocks, 256, 0, stream>>>(A, Wt, bias, degp, C,
                                                     NN, Nout, act);
    };

    const int ef_blocks = (E * 128 + 255) / 256;
    const int eh_blocks = (E * NHEADS + 255) / 256;

    // ---- pack all weights to transposed bf16 once per call ----
    pack_w_kernel<<<(128 * 128 + 255) / 256, 256, 0, stream>>>(W1, Wt1, 128);
    pack_w_kernel<<<(128 * 128 + 255) / 256, 256, 0, stream>>>(W2, Wt2, 128);
    pack_w_kernel<<<(128 * 128 + 255) / 256, 256, 0, stream>>>(Wg, Wtg, 128);
    pack_w_kernel<<<(64 * 128 + 255) / 256, 256, 0, stream>>>(Wcls, Wtc, 64);

    // ---- GCN layer 1 ----
    hipMemsetAsync(deg, 0, (size_t)NN * 4, stream);
    hipMemsetAsync(tmp, 0, (size_t)NN * HH * 4, stream);
    deg_kernel<<<(E + 255) / 256, 256, 0, stream>>>(dst, deg, E);
    scatter_feat_kernel<<<ef_blocks, 256, 0, stream>>>(x, src, dst, tmp, E);
    gemm(tmp, Wt1, b1, deg, xb, HH, 1);                // x1 = relu(agg/deg @ W1 + b1)

    // ---- GCN layer 2 ----
    hipMemsetAsync(tmp, 0, (size_t)NN * HH * 4, stream);
    scatter_feat_kernel<<<ef_blocks, 256, 0, stream>>>(xb, src, dst, tmp, E);
    gemm(tmp, Wt2, b2, deg, hb, HH, 1);                // x2 = relu(agg/deg @ W2 + b2)

    // ---- classification head ----
    gemm(hb, Wtc, bcls, nullptr, logits, CC, 0);       // logits = x2 @ Wcls + bcls
    softmax64_kernel<<<(NN + 7) / 8, 256, 0, stream>>>(logits, NN);

    // ---- GAT layer ----
    gemm(x, Wtg, nullptr, nullptr, xb, HH, 0);         // h = x @ Wg   (in xb)
    gat_es_kernel<<<(NN * NHEADS + 255) / 256, 256, 0, stream>>>(xb, a_src, a_dst,
                                                                 es, ed, NN);
    hipMemsetAsync(mm,   0, (size_t)NN * NHEADS * 4, stream);
    hipMemsetAsync(ssum, 0, (size_t)NN * NHEADS * 4, stream);
    gat_max_kernel<<<eh_blocks, 256, 0, stream>>>(es, ed, src, dst, mm, E);
    gat_exp_kernel<<<eh_blocks, 256, 0, stream>>>(es, ed, src, dst, mm, exb, ssum, E);
    hipMemsetAsync(tmp, 0, (size_t)NN * HH * 4, stream);
    gat_scatter_kernel<<<ef_blocks, 256, 0, stream>>>(xb, src, dst, exb, ssum, tmp, E);

    // ---- link prediction head ----
    link_kernel<<<(NN + 7) / 8, 256, 0, stream>>>(tmp, bg, Wl, bl, linkp, NN);
}
